// ScaledDotProductAttentionwithEdge_7834020348635
// MI455X (gfx1250) — matrix operand order repaired and checked by reading, not compiled, not run on hardware
//
#include <hip/hip_runtime.h>

typedef __attribute__((ext_vector_type(16))) _Float16 v16h;
typedef __attribute__((ext_vector_type(8)))  float    v8f;
typedef __attribute__((ext_vector_type(4)))  float    f4;
typedef __attribute__((ext_vector_type(4)))  int      i4;

#define NEG_INF_F (-1.0e9f)

__device__ __forceinline__ v8f wmma16(v16h a, v16h b, v8f c) {
  // D = A(16x32 f16) * B(32x16 f16) + C(16x16 f32)
  return __builtin_amdgcn_wmma_f32_16x16x32_f16(
      /*neg_a=*/false, a, /*neg_b=*/false, b,
      /*c_mod=*/(short)0, c, /*reuse_a=*/false, /*reuse_b=*/false);
}

// exchange value with lane^16 (combine the two half-waves): ds_swizzle SWAPX16
__device__ __forceinline__ float swap_half(float x) {
  return __builtin_bit_cast(
      float, __builtin_amdgcn_ds_swizzle(__builtin_bit_cast(int, x), 0x401F));
}

__global__ __launch_bounds__(256) void edge_attn_kernel(
    const float* __restrict__ q,     // [nodes, 64]
    const float* __restrict__ key,   // [nodes, 32, 64]
    const float* __restrict__ val,   // [nodes, 32, 64]
    const int*   __restrict__ mask,  // [nodes, 32]
    float*       __restrict__ out,   // [nodes, 64]
    int nodes) {
  const int lane = threadIdx.x & 31;
  const int wave = threadIdx.x >> 5;
  const int node = blockIdx.x * 8 + wave;    // wave-uniform
  if (node >= nodes) return;                  // whole wave exits together (EXEC stays all-1s)

  const int hi = lane >> 4;   // half-wave id (0: lanes 0-15, 1: lanes 16-31)
  const int m  = lane & 15;

  const float* qn = q    + (size_t)node * 64;
  const float* kn = key  + (size_t)node * 2048;
  const float* vn = val  + (size_t)node * 2048;
  const int*   mn = mask + (size_t)node * 32;

  // ---------- Stage 1: scores[32] = key(32x64) . q(64), via 4 WMMAs ----------
  // C layout result: lane holds v[r] = score[kc*16 + r + 8*hi], identical across
  // the 16 lanes of each half (all B columns equal).
  v8f acc0 = {};  // scores 0..15
  v8f acc1 = {};  // scores 16..31
#pragma unroll
  for (int dbase = 0; dbase < 64; dbase += 32) {
    // B operand: q replicated in every column. Lane-half holds K = 8*hi + e (e<8), 16 + 8*hi + (e-8) (e>=8).
    v16h bq;
    {
      const float* qp = qn + dbase + hi * 8;
      f4 q0 = *(const f4*)(qp + 0);
      f4 q1 = *(const f4*)(qp + 4);
      f4 q2 = *(const f4*)(qp + 16);
      f4 q3 = *(const f4*)(qp + 20);
#pragma unroll
      for (int e = 0; e < 4; ++e) {
        bq[e]      = (_Float16)q0[e];
        bq[4 + e]  = (_Float16)q1[e];
        bq[8 + e]  = (_Float16)q2[e];
        bq[12 + e] = (_Float16)q3[e];
      }
    }
    // A operand: 16 key rows x 32 d. ISA layout: lane m holds row (kc*16+m),
    // elements 0..7 -> d = dbase + 8*hi + 0..7 ; 8..15 -> d = dbase + 16 + 8*hi + 0..7
#pragma unroll
    for (int kc = 0; kc < 2; ++kc) {
      const float* kr = kn + (size_t)(kc * 16 + m) * 64 + dbase + hi * 8;
      f4 a0 = *(const f4*)(kr + 0);
      f4 a1 = *(const f4*)(kr + 4);
      f4 a2 = *(const f4*)(kr + 16);
      f4 a3 = *(const f4*)(kr + 20);
      v16h A;
#pragma unroll
      for (int e = 0; e < 4; ++e) {
        A[e]      = (_Float16)a0[e];
        A[4 + e]  = (_Float16)a1[e];
        A[8 + e]  = (_Float16)a2[e];
        A[12 + e] = (_Float16)a3[e];
      }
      if (kc == 0) acc0 = wmma16(A, bq, acc0);
      else         acc1 = wmma16(A, bq, acc1);
    }
  }

  // ---------- Softmax over 32 scores (f32), halves combined via ds_swizzle ----------
  // This lane's k indices: acc0[r] -> k = 8*hi + r ; acc1[r] -> k = 16 + 8*hi + r
  i4 mk0 = *(const i4*)(mn + 8 * hi);
  i4 mk1 = *(const i4*)(mn + 8 * hi + 4);
  i4 mk2 = *(const i4*)(mn + 16 + 8 * hi);
  i4 mk3 = *(const i4*)(mn + 16 + 8 * hi + 4);

  const float scale = 0.125f;  // 1/sqrt(64)
  float s0[8], s1[8];
#pragma unroll
  for (int r = 0; r < 4; ++r) {
    s0[r]     = mk0[r] ? acc0[r]     * scale : NEG_INF_F;
    s0[4 + r] = mk1[r] ? acc0[4 + r] * scale : NEG_INF_F;
    s1[r]     = mk2[r] ? acc1[r]     * scale : NEG_INF_F;
    s1[4 + r] = mk3[r] ? acc1[4 + r] * scale : NEG_INF_F;
  }
  float mx = s0[0];
#pragma unroll
  for (int r = 1; r < 8; ++r) mx = fmaxf(mx, s0[r]);
#pragma unroll
  for (int r = 0; r < 8; ++r) mx = fmaxf(mx, s1[r]);
  mx = fmaxf(mx, swap_half(mx));  // global max across all 32 k

  float p0[8], p1[8], sum = 0.0f;
#pragma unroll
  for (int r = 0; r < 8; ++r) { p0[r] = __expf(s0[r] - mx); sum += p0[r]; }
#pragma unroll
  for (int r = 0; r < 8; ++r) { p1[r] = __expf(s1[r] - mx); sum += p1[r]; }
  sum += swap_half(sum);          // global sum across all 32 k

  // ---------- Stage 2: O[64] = attn(1x32) . V(32x64), via 4 WMMAs ----------
  // A operand = attn replicated in every row. Required layout: lane needs
  // K = 8*hi + 0..7 (elems 0..7) and K = 16 + 8*hi + 0..7 (elems 8..15) —
  // exactly what this lane already holds in p0/p1. No cross-lane traffic.
  v16h A2;
#pragma unroll
  for (int r = 0; r < 8; ++r) {
    A2[r]     = (_Float16)p0[r];
    A2[8 + r] = (_Float16)p1[r];
  }

  float o0, o1, o2, o3;
#pragma unroll
  for (int c = 0; c < 4; ++c) {
    // B operand: V column chunk. Lane j holds V[K, 16c + j], K = 8*hi + e (e<8), 16 + 8*hi + (e-8) (e>=8).
    const float* vp = vn + (size_t)(hi * 8) * 64 + 16 * c + m;
    v16h B2;
#pragma unroll
    for (int r = 0; r < 16; ++r) B2[r] = (_Float16)vp[(size_t)(r + (r >= 8 ? 8 : 0)) * 64];
    v8f z = {};
    v8f C2 = wmma16(A2, B2, z);
    float oc = C2[0];  // all rows equal -> any accumulator register
    if      (c == 0) o0 = oc;
    else if (c == 1) o1 = oc;
    else if (c == 2) o2 = oc;
    else             o3 = oc;
  }

  // ---------- Store: two fully-coalesced 128B stores per node ----------
  const float inv = 1.0f / sum;   // softmax normalization folded in here
  float* op = out + (size_t)node * 64;
  float v0 = hi ? o1 : o0;        // chunk c = hi
  float v1 = hi ? o3 : o2;        // chunk c = 2 + hi
  op[16 * hi + m]      = v0 * inv;
  op[32 + 16 * hi + m] = v1 * inv;
}

extern "C" void kernel_launch(void* const* d_in, const int* in_sizes, int n_in,
                              void* d_out, int out_size, void* d_ws, size_t ws_size,
                              hipStream_t stream) {
  const float* q  = (const float*)d_in[0];  // [B,H,N,D]
  const float* k  = (const float*)d_in[1];  // [B,H,N,K,D]
  const float* v  = (const float*)d_in[2];  // [B,H,N,K,D]
  const int*   mk = (const int*)d_in[3];    // [B,H,N,K]
  float* out = (float*)d_out;               // [B,H,N,D]

  const int nodes  = in_sizes[0] / 64;      // B*H*N
  const int blocks = (nodes + 7) / 8;       // 8 waves (nodes) per 256-thread block
  edge_attn_kernel<<<blocks, 256, 0, stream>>>(q, k, v, mk, out, nodes);
}
